// OfficialAKOrN_78271484003251
// MI455X (gfx1250) — matrix operand order repaired and checked by reading, not compiled, not run on hardware
//
#include <hip/hip_runtime.h>
#include <hip/hip_bf16.h>

typedef __attribute__((ext_vector_type(16))) _Float16 v16h;
typedef __attribute__((ext_vector_type(8)))  _Float16 v8h;
typedef __attribute__((ext_vector_type(8)))  float    v8f;

#define WMMA_F16(a, b, c) \
  __builtin_amdgcn_wmma_f32_16x16x32_f16(false, (a), false, (b), (short)0, (c), false, false)

// ---------- problem constants ----------
#define BB     8
#define NTOK   1024
#define DCH    512
#define HEADS  8
#define HD     64
#define TSTEPS 10
#define NGRP   (BB * NTOK * (DCH / 4))   // 1,048,576 four-dim oscillator groups
#define MROWS  (BB * NTOK)               // 8192 tokens

// ---------- WMMA fragment loaders (ISA 7.12.2 layouts) ----------
__device__ __forceinline__ v8f vzero8() {
  v8f z = {0.f, 0.f, 0.f, 0.f, 0.f, 0.f, 0.f, 0.f};
  return z;
}

// A-matrix 16x32 f16, row-major source with leading dim lda (in halfs).
// lane l: row m = l&15 ; halfs 0..7 = K k0+kg..+7, halfs 8..15 = K k0+16+kg..+7, kg = 8*(l>>4)
__device__ __forceinline__ v16h load_afrag(const _Float16* A, int lda, int k0, int lane) {
  const int m  = lane & 15;
  const int kg = (lane >> 4) << 3;
  const _Float16* p = A + (size_t)m * lda + k0 + kg;
  v8h lo = *(const v8h*)(p);
  v8h hi = *(const v8h*)(p + 16);
  return __builtin_shufflevector(lo, hi, 0, 1, 2, 3, 4, 5, 6, 7, 8, 9, 10, 11, 12, 13, 14, 15);
}

// B-matrix 32x16 f16, row-major source (K-major): lane l holds col c0+(l&15); halfs 0..7 = K k0+kg..+7, halfs 8..15 = K k0+16+kg..+7, kg = 8*(l>>4)
__device__ __forceinline__ v16h load_bfrag(const _Float16* B, int ldb, int k0, int c0, int lane) {
  const int n  = lane & 15;
  const int kg = (lane >> 4) << 3;
  const _Float16* p = B + (size_t)(k0 + kg) * ldb + c0 + n;
  v16h r;
#pragma unroll
  for (int s = 0; s < 8; ++s) {
    r[s]     = p[(size_t)s * ldb];
    r[s + 8] = p[(size_t)(16 + s) * ldb];
  }
  return r;
}

// 32x32 output tile main loop: acc4[mi*2+ni] over K
__device__ __forceinline__ void gemm_tile(const _Float16* __restrict__ A, int lda,
                                          const _Float16* __restrict__ Bw, int ldb,
                                          int r0, int c0, int K, int lane, v8f* acc4) {
  const _Float16* A0 = A + (size_t)r0 * lda;
  const _Float16* A1 = A + (size_t)(r0 + 16) * lda;
  for (int k0 = 0; k0 < K; k0 += 32) {
    v16h a0 = load_afrag(A0, lda, k0, lane);
    v16h a1 = load_afrag(A1, lda, k0, lane);
    v16h b0 = load_bfrag(Bw, ldb, k0, c0, lane);
    v16h b1 = load_bfrag(Bw, ldb, k0, c0 + 16, lane);
    acc4[0] = WMMA_F16(a0, b0, acc4[0]);
    acc4[1] = WMMA_F16(a0, b1, acc4[1]);
    acc4[2] = WMMA_F16(a1, b0, acc4[2]);
    acc4[3] = WMMA_F16(a1, b1, acc4[3]);
  }
}

// ---------- elementwise / prep kernels ----------
__global__ void transpose_w_f16_kernel(const float* __restrict__ W, _Float16* __restrict__ WT,
                                       int R, int K) {
  int i = blockIdx.x * blockDim.x + threadIdx.x;
  if (i >= R * K) return;
  int r = i / K, k = i - r * K;
  WT[(size_t)k * R + r] = (_Float16)W[i];
}

__global__ void convert_f16_kernel(const float* __restrict__ src, _Float16* __restrict__ dst, int n) {
  int i = blockIdx.x * blockDim.x + threadIdx.x;
  if (i < n) dst[i] = (_Float16)src[i];
}

// zt = z / max(||z_group||, 1e-6) per 4-dim chunk; writes f32 state + f16 copy
__global__ void knorm_init_kernel(const float* __restrict__ z, float* __restrict__ xt,
                                  _Float16* __restrict__ xth) {
  int i = blockIdx.x * blockDim.x + threadIdx.x;
  if (i >= NGRP) return;
  size_t b = (size_t)i * 4;
  float a0 = z[b], a1 = z[b + 1], a2 = z[b + 2], a3 = z[b + 3];
  float inv = 1.f / fmaxf(sqrtf(a0 * a0 + a1 * a1 + a2 * a2 + a3 * a3), 1e-6f);
  float u0 = a0 * inv, u1 = a1 * inv, u2 = a2 * inv, u3 = a3 * inv;
  xt[b] = u0; xt[b + 1] = u1; xt[b + 2] = u2; xt[b + 3] = u3;
  xth[b] = (_Float16)u0; xth[b + 1] = (_Float16)u1; xth[b + 2] = (_Float16)u2; xth[b + 3] = (_Float16)u3;
}

// xt <- knorm(xt + (y - <xt,y> xt)) per 4-group  (GAMMA = 1)
__global__ void update_kernel(const float* __restrict__ y, float* __restrict__ xt,
                              _Float16* __restrict__ xth) {
  int i = blockIdx.x * blockDim.x + threadIdx.x;
  if (i >= NGRP) return;
  size_t b = (size_t)i * 4;
  float x0 = xt[b], x1 = xt[b + 1], x2 = xt[b + 2], x3 = xt[b + 3];
  float y0 = y[b], y1 = y[b + 1], y2 = y[b + 2], y3 = y[b + 3];
  float sim = x0 * y0 + x1 * y1 + x2 * y2 + x3 * y3;
  float u0 = x0 + (y0 - sim * x0);
  float u1 = x1 + (y1 - sim * x1);
  float u2 = x2 + (y2 - sim * x2);
  float u3 = x3 + (y3 - sim * x3);
  float inv = 1.f / fmaxf(sqrtf(u0 * u0 + u1 * u1 + u2 * u2 + u3 * u3), 1e-6f);
  u0 *= inv; u1 *= inv; u2 *= inv; u3 *= inv;
  xt[b] = u0; xt[b + 1] = u1; xt[b + 2] = u2; xt[b + 3] = u3;
  xth[b] = (_Float16)u0; xth[b + 1] = (_Float16)u1; xth[b + 2] = (_Float16)u2; xth[b + 3] = (_Float16)u3;
}

// GroupNorm over (N tokens x 4 chans) per (batch, group-of-4). 1024 blocks.
__global__ __launch_bounds__(256) void groupnorm_kernel(const float* __restrict__ craw,
                                                        const float* __restrict__ gw,
                                                        const float* __restrict__ gb,
                                                        float* __restrict__ c) {
  __shared__ float ssum[256], ssq[256];
  const int b = blockIdx.x >> 7, g = blockIdx.x & 127;
  const int tid = threadIdx.x;
  float s = 0.f, q = 0.f;
  for (int i = tid; i < 4096; i += 256) {
    int nn = i >> 2, r = i & 3;
    float v = craw[((size_t)(b * NTOK + nn)) * DCH + g * 4 + r];
    s += v; q += v * v;
  }
  ssum[tid] = s; ssq[tid] = q;
  __syncthreads();
  for (int st = 128; st > 0; st >>= 1) {
    if (tid < st) { ssum[tid] += ssum[tid + st]; ssq[tid] += ssq[tid + st]; }
    __syncthreads();
  }
  float mean = ssum[0] * (1.f / 4096.f);
  float var  = ssq[0] * (1.f / 4096.f) - mean * mean;
  float rstd = rsqrtf(var + 1e-5f);
  for (int i = tid; i < 4096; i += 256) {
    int nn = i >> 2, r = i & 3;
    int ch = g * 4 + r;
    size_t idx = ((size_t)(b * NTOK + nn)) * DCH + ch;
    c[idx] = (craw[idx] - mean) * rstd * gw[ch] + gb[ch];
  }
}

// ---------- GEMM kernels (WMMA) ----------
// Generic: C_f32[M,N] = A_f16[M,K] * Bw_f16[K,N] + bias[N] (+ addend[M,N] if non-null)
__global__ __launch_bounds__(256) void gemm_bias_add_f32_kernel(
    const _Float16* __restrict__ A, const _Float16* __restrict__ Bw,
    const float* __restrict__ bias, const float* __restrict__ addend,
    float* __restrict__ C, int M, int N, int K) {
  const int lane = threadIdx.x & 31, w = threadIdx.x >> 5;
  const int r0 = blockIdx.y * 128 + (w & 3) * 32;
  const int c0 = blockIdx.x * 64 + (w >> 2) * 32;
  v8f acc4[4];
  acc4[0] = vzero8(); acc4[1] = vzero8(); acc4[2] = vzero8(); acc4[3] = vzero8();
  gemm_tile(A, K, Bw, N, r0, c0, K, lane, acc4);
  const int n16 = lane & 15, vh = (lane >> 4) * 8;
#pragma unroll
  for (int mi = 0; mi < 2; ++mi)
#pragma unroll
    for (int ni = 0; ni < 2; ++ni)
#pragma unroll
      for (int v = 0; v < 8; ++v) {
        int row = r0 + mi * 16 + vh + v;
        int col = c0 + ni * 16 + n16;
        size_t idx = (size_t)row * N + col;
        float val = acc4[mi * 2 + ni][v] + bias[col];
        if (addend) val += addend[idx];
        C[idx] = val;
      }
}

// qkv GEMM: N = 1536; epilogue scatters into q[b,h,n,d], kT[b,h,d,n], v[b,h,n,d] (f16)
__global__ __launch_bounds__(256) void gemm_qkv_kernel(
    const _Float16* __restrict__ A, const _Float16* __restrict__ Bw,
    const float* __restrict__ bias, _Float16* __restrict__ q,
    _Float16* __restrict__ kT, _Float16* __restrict__ vb) {
  const int lane = threadIdx.x & 31, w = threadIdx.x >> 5;
  const int r0 = blockIdx.y * 128 + (w & 3) * 32;
  const int c0 = blockIdx.x * 64 + (w >> 2) * 32;
  v8f acc4[4];
  acc4[0] = vzero8(); acc4[1] = vzero8(); acc4[2] = vzero8(); acc4[3] = vzero8();
  gemm_tile(A, DCH, Bw, 3 * DCH, r0, c0, DCH, lane, acc4);
  const int n16 = lane & 15, vh = (lane >> 4) * 8;
#pragma unroll
  for (int mi = 0; mi < 2; ++mi)
#pragma unroll
    for (int ni = 0; ni < 2; ++ni)
#pragma unroll
      for (int v = 0; v < 8; ++v) {
        int tok  = r0 + mi * 16 + vh + v;
        int chan = c0 + ni * 16 + n16;
        float val = acc4[mi * 2 + ni][v] + bias[chan];
        _Float16 hv = (_Float16)val;
        int sec = chan / DCH, c5 = chan - sec * DCH;
        int hh = c5 >> 6, dd = c5 & 63;
        int bb = tok >> 10, nn = tok & (NTOK - 1);
        size_t bh = (size_t)(bb * HEADS + hh);
        if (sec == 0)      q[(bh * NTOK + nn) * HD + dd] = hv;
        else if (sec == 2) vb[(bh * NTOK + nn) * HD + dd] = hv;
        else               kT[(bh * HD + dd) * NTOK + nn] = hv;
      }
}

// readout GEMM: v = z*WinvT + binv (N=2048); fused group-of-4 L2 norm + ro_bias -> out f32
__global__ __launch_bounds__(256) void gemm_readout_kernel(
    const _Float16* __restrict__ A, const _Float16* __restrict__ Bw,
    const float* __restrict__ binv, const float* __restrict__ ro_bias,
    float* __restrict__ out) {
  const int lane = threadIdx.x & 31, w = threadIdx.x >> 5;
  const int r0 = blockIdx.y * 128 + (w & 3) * 32;
  const int c0 = blockIdx.x * 64 + (w >> 2) * 32;
  v8f acc4[4];
  acc4[0] = vzero8(); acc4[1] = vzero8(); acc4[2] = vzero8(); acc4[3] = vzero8();
  gemm_tile(A, DCH, Bw, 4 * DCH, r0, c0, DCH, lane, acc4);
  const int n16 = lane & 15, vh = (lane >> 4) * 8;
#pragma unroll
  for (int mi = 0; mi < 2; ++mi)
#pragma unroll
    for (int ni = 0; ni < 2; ++ni)
#pragma unroll
      for (int v = 0; v < 8; ++v) {
        int row = r0 + mi * 16 + vh + v;
        int col = c0 + ni * 16 + n16;
        float t = acc4[mi * 2 + ni][v] + binv[col];
        float s = t * t;
        s += __shfl_xor(s, 1);
        s += __shfl_xor(s, 2);
        if ((lane & 3) == 0) {
          int oc = col >> 2;
          out[(size_t)row * DCH + oc] = sqrtf(s) + ro_bias[oc];
        }
      }
}

// ---------- flash attention, one wave per 16-query block ----------
__global__ __launch_bounds__(32) void attn_kernel(const _Float16* __restrict__ q,
                                                  const _Float16* __restrict__ kT,
                                                  const _Float16* __restrict__ vv,
                                                  _Float16* __restrict__ o) {
  const int lane = threadIdx.x;
  const int qb = blockIdx.x & 63;          // N/16 blocks
  const int h  = (blockIdx.x >> 6) & 7;
  const int b  = blockIdx.x >> 9;
  const size_t bh = (size_t)(b * HEADS + h);
  const _Float16* qp = q + (bh * NTOK + qb * 16) * HD;
  const _Float16* kp = kT + bh * HD * NTOK;   // [HD][NTOK]
  const _Float16* vp = vv + bh * NTOK * HD;   // [NTOK][HD]

  const v16h qa0 = load_afrag(qp, HD, 0, lane);
  const v16h qa1 = load_afrag(qp, HD, 32, lane);

  v8f oacc[4];
  oacc[0] = vzero8(); oacc[1] = vzero8(); oacc[2] = vzero8(); oacc[3] = vzero8();
  float rm[8], rs[8];
#pragma unroll
  for (int i = 0; i < 8; ++i) { rm[i] = -1e30f; rs[i] = 0.f; }

  __shared__ __align__(32) _Float16 pl[16 * 32];
  const float scale = 0.125f;               // 1/sqrt(64)
  const int n16 = lane & 15, hf = lane >> 4;

  for (int j0 = 0; j0 < NTOK; j0 += 32) {
    v16h kb00 = load_bfrag(kp, NTOK, 0, j0, lane);       // d 0..31, keys j0..+15
    v16h kb01 = load_bfrag(kp, NTOK, 32, j0, lane);      // d 32..63
    v16h kb10 = load_bfrag(kp, NTOK, 0, j0 + 16, lane);  // keys j0+16..+31
    v16h kb11 = load_bfrag(kp, NTOK, 32, j0 + 16, lane);
    v8f s0 = vzero8(), s1 = vzero8();
    s0 = WMMA_F16(qa0, kb00, s0); s0 = WMMA_F16(qa1, kb01, s0);
    s1 = WMMA_F16(qa0, kb10, s1); s1 = WMMA_F16(qa1, kb11, s1);

    float al[8];
#pragma unroll
    for (int v = 0; v < 8; ++v) {
      float a0 = s0[v] * scale, a1 = s1[v] * scale;
      float m = fmaxf(a0, a1);
      m = fmaxf(m, __shfl_xor(m, 1));
      m = fmaxf(m, __shfl_xor(m, 2));
      m = fmaxf(m, __shfl_xor(m, 4));
      m = fmaxf(m, __shfl_xor(m, 8));
      float nm = fmaxf(rm[v], m);
      al[v] = __expf(rm[v] - nm);
      float p0 = __expf(a0 - nm), p1 = __expf(a1 - nm);
      int row = v + hf * 8;
      pl[row * 32 + n16] = (_Float16)p0;
      pl[row * 32 + 16 + n16] = (_Float16)p1;
      float ps = p0 + p1;
      ps += __shfl_xor(ps, 1);
      ps += __shfl_xor(ps, 2);
      ps += __shfl_xor(ps, 4);
      ps += __shfl_xor(ps, 8);
      rs[v] = rs[v] * al[v] + ps;
      rm[v] = nm;
    }
    __syncthreads();
    v16h pa = load_afrag(pl, 32, 0, lane);   // P as 16x32 A-frag
#pragma unroll
    for (int t = 0; t < 4; ++t) {
      v16h vb = load_bfrag(vp, HD, j0, t * 16, lane);
#pragma unroll
      for (int v = 0; v < 8; ++v) oacc[t][v] *= al[v];
      oacc[t] = WMMA_F16(pa, vb, oacc[t]);
    }
    __syncthreads();
  }
#pragma unroll
  for (int t = 0; t < 4; ++t)
#pragma unroll
    for (int v = 0; v < 8; ++v) {
      int tok  = b * NTOK + qb * 16 + v + hf * 8;
      int chan = h * HD + t * 16 + n16;
      o[(size_t)tok * DCH + chan] = (_Float16)(oacc[t][v] / rs[v]);
    }
}

// ---------- launch ----------
extern "C" void kernel_launch(void* const* d_in, const int* in_sizes, int n_in,
                              void* d_out, int out_size, void* d_ws, size_t ws_size,
                              hipStream_t stream) {
  (void)in_sizes; (void)n_in; (void)out_size; (void)ws_size;
  const float* x     = (const float*)d_in[0];
  const float* z     = (const float*)d_in[1];
  const float* Wc    = (const float*)d_in[2];
  const float* bc    = (const float*)d_in[3];
  const float* gn_w  = (const float*)d_in[4];
  const float* gn_b  = (const float*)d_in[5];
  const float* Wqkv  = (const float*)d_in[6];
  const float* bqkv  = (const float*)d_in[7];
  const float* Wo    = (const float*)d_in[8];
  const float* bo    = (const float*)d_in[9];
  const float* Winv  = (const float*)d_in[10];
  const float* binv  = (const float*)d_in[11];
  const float* ro_b  = (const float*)d_in[12];
  float* out = (float*)d_out;

  char* base = (char*)d_ws;
  size_t off = 0;
  auto take = [&](size_t bytes) -> char* {
    char* p = base + off;
    off += (bytes + 255) & ~(size_t)255;
    return p;
  };
  _Float16* WcT   = (_Float16*)take((size_t)DCH * DCH * 2);
  _Float16* WqkvT = (_Float16*)take((size_t)DCH * 3 * DCH * 2);
  _Float16* WoT   = (_Float16*)take((size_t)DCH * DCH * 2);
  _Float16* WinvT = (_Float16*)take((size_t)DCH * 4 * DCH * 2);
  float*    cbuf  = (float*)take((size_t)MROWS * DCH * 4);
  float*    ybuf  = (float*)take((size_t)MROWS * DCH * 4);
  float*    xt    = (float*)take((size_t)MROWS * DCH * 4);
  _Float16* xth   = (_Float16*)take((size_t)MROWS * DCH * 2);
  _Float16* qb    = (_Float16*)take((size_t)MROWS * DCH * 2);
  _Float16* kTb   = (_Float16*)take((size_t)MROWS * DCH * 2);
  _Float16* vb    = (_Float16*)take((size_t)MROWS * DCH * 2);
  _Float16* ob    = (_Float16*)take((size_t)MROWS * DCH * 2);

  // 1) weights -> K-major f16
  transpose_w_f16_kernel<<<(DCH * DCH + 255) / 256, 256, 0, stream>>>(Wc, WcT, DCH, DCH);
  transpose_w_f16_kernel<<<(3 * DCH * DCH + 255) / 256, 256, 0, stream>>>(Wqkv, WqkvT, 3 * DCH, DCH);
  transpose_w_f16_kernel<<<(DCH * DCH + 255) / 256, 256, 0, stream>>>(Wo, WoT, DCH, DCH);
  transpose_w_f16_kernel<<<(4 * DCH * DCH + 255) / 256, 256, 0, stream>>>(Winv, WinvT, 4 * DCH, DCH);

  // 2) conditioning: craw = x*WcT + bc (reuse ybuf), then GroupNorm -> cbuf
  convert_f16_kernel<<<(MROWS * DCH + 255) / 256, 256, 0, stream>>>(x, xth, MROWS * DCH);
  gemm_bias_add_f32_kernel<<<dim3(DCH / 64, MROWS / 128), 256, 0, stream>>>(
      xth, WcT, bc, nullptr, ybuf, MROWS, DCH, DCH);
  groupnorm_kernel<<<BB * 128, 256, 0, stream>>>(ybuf, gn_w, gn_b, cbuf);

  // 3) initial oscillator state on the hypersphere
  knorm_init_kernel<<<NGRP / 256, 256, 0, stream>>>(z, xt, xth);

  // 4) Kuramoto iterations
  for (int t = 0; t < TSTEPS; ++t) {
    gemm_qkv_kernel<<<dim3(3 * DCH / 64, MROWS / 128), 256, 0, stream>>>(
        xth, WqkvT, bqkv, qb, kTb, vb);
    attn_kernel<<<BB * HEADS * (NTOK / 16), 32, 0, stream>>>(qb, kTb, vb, ob);
    gemm_bias_add_f32_kernel<<<dim3(DCH / 64, MROWS / 128), 256, 0, stream>>>(
        ob, WoT, bo, cbuf, ybuf, MROWS, DCH, DCH);
    update_kernel<<<NGRP / 256, 256, 0, stream>>>(ybuf, xt, xth);
  }

  // 5) readout with fused rotation-invariant norm
  gemm_readout_kernel<<<dim3(4 * DCH / 64, MROWS / 128), 256, 0, stream>>>(
      xth, WinvT, binv, ro_b, out);
}
